// CAPE_Decoder_no_pose_80341658239447
// MI455X (gfx1250) — matrix operand order repaired and checked by reading, not compiled, not run on hardware
//
#include <hip/hip_runtime.h>
#include <hip/hip_bf16.h>

// ---------------------------------------------------------------------------
// CAPE decoder pipeline for MI455X (gfx1250, wave32, WMMA).
// GEMM core: v_wmma_f32_16x16x32_bf16 (bf16 in / f32 accum). A block owns a
// 128x32 macro-tile: 8 waves stacked in M share one 32-column tile, so the
// 32(K)x32(N) weight slab is staged in LDS once per block per K-step via
// GLOBAL_LOAD_ASYNC_TO_LDS_B128 (ASYNCcnt + s_wait_asynccnt + barrier),
// cutting B L2 traffic 8x. Epilogue variants are compile-time template
// parameters (branch-free stores). Graph propagation = COO scatter via
// global_atomic_add_f32. GroupNorm = LDS tree reduction.
// ---------------------------------------------------------------------------

typedef __attribute__((ext_vector_type(16))) __bf16 v16bf;
typedef __attribute__((ext_vector_type(8)))  float  v8f;

#define BSZ 32          // batch size
#define GEMM_WAVES 8    // waves per block (stacked along M)
#define ASYNC_STAGE 1   // use GLOBAL_LOAD_ASYNC_TO_LDS_B128 for W staging

typedef __attribute__((address_space(3))) float lds_float;

// ------------------------------ WMMA GEMM ----------------------------------
// Out[M,N] = epilogue( A[M,K] @ W[K,N] )   (row-major, fp32 in memory)
// Requires M%16==0, N%32==0, K%32==0 (true for every GEMM in this net).
template<bool HAS_BIAS, bool HAS_CIN, bool LEAKY>
__global__ void __launch_bounds__(256)
gemm_bf16_wmma(const float* __restrict__ A, const float* __restrict__ W,
               const float* __restrict__ Cin, const float* __restrict__ bias,
               float* __restrict__ Out, int M, int K, int N)
{
    const int lane   = threadIdx.x & 31;
    const int wave   = threadIdx.x >> 5;
    const int tilesN = N >> 5;                       // 32-wide column tiles
    const int mTiles = M >> 4;
    const int bn  = blockIdx.x % tilesN;             // column tile (block-wide)
    const int tm  = (blockIdx.x / tilesN) * GEMM_WAVES + wave;
    const bool valid = tm < mTiles;                  // wave-uniform => EXEC full

    const int  lo = lane & 15;
    const bool hi = lane >= 16;
    const int  koff = hi ? 16 : 0;

    // A fragment (16x32 bf16): lane holds row (tm*16+lo); per K-step the lane
    // needs K = k0 + {0|8} + [0..7]  and  K = k0 + 16 + {0|8} + [0..7].
    const float* arow = A + (size_t)((valid ? tm : 0) * 16 + lo) * K + (hi ? 8 : 0);

    // --- W slab staging: 32(K) x 32(N) f32 in LDS, one async B128 / thread ---
    __shared__ float tileW[32 * 32];
    const int stIdx = threadIdx.x * 4;               // element idx, 16B aligned
    const int stKr  = stIdx >> 5;                    // K row within slab
    const int stC   = stIdx & 31;                    // column within slab
    const float* gW = W + bn * 32 + stC;             // global column base
#if ASYNC_STAGE
    const unsigned ldsOff = (unsigned)(size_t)(lds_float*)&tileW[stIdx];
#endif

    v8f acc0 = {0.f,0.f,0.f,0.f,0.f,0.f,0.f,0.f};
    v8f acc1 = {0.f,0.f,0.f,0.f,0.f,0.f,0.f,0.f};

    for (int k0 = 0; k0 < K; k0 += 32) {
        __syncthreads();                             // previous reads done
#if ASYNC_STAGE
        asm volatile("global_load_async_to_lds_b128 %0, %1, off"
                     :: "v"(ldsOff), "v"(gW + (size_t)(k0 + stKr) * N)
                     : "memory");
        asm volatile("s_wait_asynccnt 0x0" ::: "memory");
#else
        *(float4*)&tileW[stIdx] = *(const float4*)(gW + (size_t)(k0 + stKr) * N);
#endif
        __syncthreads();                             // slab visible to all

        if (valid) {
            if (k0 + 32 < K) __builtin_prefetch(arow + k0 + 32, 0, 1);

            float4 a0 = *(const float4*)(arow + k0);
            float4 a1 = *(const float4*)(arow + k0 + 4);
            float4 a2 = *(const float4*)(arow + k0 + 16);
            float4 a3 = *(const float4*)(arow + k0 + 20);
            v16bf av;
            av[0]=(__bf16)a0.x;  av[1]=(__bf16)a0.y;  av[2]=(__bf16)a0.z;  av[3]=(__bf16)a0.w;
            av[4]=(__bf16)a1.x;  av[5]=(__bf16)a1.y;  av[6]=(__bf16)a1.z;  av[7]=(__bf16)a1.w;
            av[8]=(__bf16)a2.x;  av[9]=(__bf16)a2.y;  av[10]=(__bf16)a2.z; av[11]=(__bf16)a2.w;
            av[12]=(__bf16)a3.x; av[13]=(__bf16)a3.y; av[14]=(__bf16)a3.z; av[15]=(__bf16)a3.w;

            // B fragments from LDS: col lo / lo+16, K rows {0|8}+[0..7] and 16+{0|8}+[0..7]
            v16bf bv0, bv1;
#pragma unroll
            for (int i = 0; i < 16; ++i) {
                bv0[i] = (__bf16)tileW[((koff >> 1) + (i & 7) + ((i & 8) << 1)) * 32 + lo];
                bv1[i] = (__bf16)tileW[((koff >> 1) + (i & 7) + ((i & 8) << 1)) * 32 + lo + 16];
            }

            acc0 = __builtin_amdgcn_wmma_f32_16x16x32_bf16(
                       false, av, false, bv0, (short)0, acc0, false, false);
            acc1 = __builtin_amdgcn_wmma_f32_16x16x32_bf16(
                       false, av, false, bv1, (short)0, acc1, false, false);
        }
    }

    if (!valid) return;

    // C/D layout: VGPR r -> row = r + 8*(lane>=16), col = lane&15
    const int colA = bn * 32 + lo;
    float b0 = 0.f, b1 = 0.f;
    if (HAS_BIAS) { b0 = bias[colA]; b1 = bias[colA + 16]; }

#pragma unroll
    for (int r = 0; r < 8; ++r) {
        const int row = tm * 16 + r + (hi ? 8 : 0);
        const size_t o = (size_t)row * N + colA;
        float v0 = acc0[r];
        float v1 = acc1[r];
        if (HAS_BIAS) { v0 += b0;          v1 += b1; }
        if (HAS_CIN)  { v0 += Cin[o];      v1 += Cin[o + 16]; }
        if (LEAKY)    { v0 = v0 > 0.f ? v0 : 0.2f * v0;
                        v1 = v1 > 0.f ? v1 : 0.2f * v1; }
        Out[o]      = v0;
        Out[o + 16] = v1;
    }
}

// ------------------------------ zero fill ----------------------------------
__global__ void zero_kernel(float* __restrict__ p, long long n)
{
    long long i = (long long)blockIdx.x * blockDim.x + threadIdx.x;
    const long long s = (long long)gridDim.x * blockDim.x;
    for (; i < n; i += s) p[i] = 0.f;
}

// -------------------------- graph propagation ------------------------------
// Out[b,dst,c] += X[b,src,c] * norm[e]   (COO scatter, Out pre-zeroed)
__global__ void prop_kernel(const float* __restrict__ X, float* __restrict__ Out,
                            const int* __restrict__ src, const int* __restrict__ dst,
                            const float* __restrict__ nrm, int E, int Nn, int C)
{
    long long t = (long long)blockIdx.x * blockDim.x + threadIdx.x;
    if (t >= (long long)E * C) return;
    const int e = (int)(t / C);
    const int c = (int)(t - (long long)e * C);
    const int s = src[e], d = dst[e];
    const float w = nrm[e];
    const size_t sb = (size_t)s * C + c;
    const size_t db = (size_t)d * C + c;
    const size_t strd = (size_t)Nn * C;
#pragma unroll 4
    for (int b = 0; b < BSZ; ++b)
        atomicAdd(Out + db + (size_t)b * strd, X[sb + (size_t)b * strd] * w);
}

// ------------------------------- pool --------------------------------------
// Out[b,n,c] = sum_k X[b, cols[n,k], c] * vals[n,k]
__global__ void pool_kernel(const float* __restrict__ X, const int* __restrict__ cols,
                            const float* __restrict__ vals, float* __restrict__ Out,
                            int nu, int nd, int C)
{
    long long t = (long long)blockIdx.x * blockDim.x + threadIdx.x;
    if (t >= (long long)BSZ * nu * C) return;
    const int c = (int)(t % C);
    const long long nn = t / C;
    const int n = (int)(nn % nu);
    const int b = (int)(nn / nu);
    const int*   cr = cols + 3 * n;
    const float* vr = vals + 3 * n;
    const float* xb = X + (size_t)b * nd * C;
    Out[t] = xb[(size_t)cr[0] * C + c] * vr[0]
           + xb[(size_t)cr[1] * C + c] * vr[1]
           + xb[(size_t)cr[2] * C + c] * vr[2];
}

// ---------------------- GroupNorm (32 groups) + ReLU -----------------------
// one block per (batch, group); LDS tree reduction for mean/var
__global__ void __launch_bounds__(256)
gn_relu_kernel(const float* __restrict__ X, const float* __restrict__ scale,
               const float* __restrict__ bias, float* __restrict__ Out,
               int Nn, int C)
{
    const int b  = blockIdx.x >> 5;
    const int g  = blockIdx.x & 31;
    const int cg = C >> 5;
    const int cnt = Nn * cg;
    const float* base  = X   + (size_t)b * Nn * C + g * cg;
    float*       obase = Out + (size_t)b * Nn * C + g * cg;

    float s = 0.f, q = 0.f;
    for (int i = threadIdx.x; i < cnt; i += 256) {
        const int n = i / cg, j = i - n * cg;
        const float v = base[(size_t)n * C + j];
        s += v; q += v * v;
    }
    __shared__ float shs[256], shq[256];
    shs[threadIdx.x] = s; shq[threadIdx.x] = q;
    __syncthreads();
    for (int w = 128; w > 0; w >>= 1) {
        if (threadIdx.x < w) {
            shs[threadIdx.x] += shs[threadIdx.x + w];
            shq[threadIdx.x] += shq[threadIdx.x + w];
        }
        __syncthreads();
    }
    const float mean = shs[0] / (float)cnt;
    const float var  = shq[0] / (float)cnt - mean * mean;
    const float rstd = rsqrtf(var + 1e-5f);

    for (int i = threadIdx.x; i < cnt; i += 256) {
        const int n = i / cg, j = i - n * cg;
        const size_t o = (size_t)n * C + j;
        float v = (base[o] - mean) * rstd * scale[g * cg + j] + bias[g * cg + j];
        obase[o] = fmaxf(v, 0.f);
    }
}

// ---------------------- final K=2 conv to 3 channels -----------------------
// Out[b,n,:] = H[b,n,:] @ W[0] + P[b,n,:] @ W[1] + bias[n,:]
__global__ void __launch_bounds__(256)
conv_out_kernel(const float* __restrict__ H, const float* __restrict__ P,
                const float* __restrict__ W, const float* __restrict__ bias,
                float* __restrict__ Out, int Nn)
{
    __shared__ float sw[384];                    // (2,64,3) weights in LDS
    for (int i = threadIdx.x; i < 384; i += 256) sw[i] = W[i];
    __syncthreads();

    const int t = blockIdx.x * 256 + threadIdx.x;
    if (t >= BSZ * Nn) return;
    const int n = t % Nn;
    const float* h = H + (size_t)t * 64;
    const float* p = P + (size_t)t * 64;
    float o0 = 0.f, o1 = 0.f, o2 = 0.f;
#pragma unroll 8
    for (int c = 0; c < 64; ++c) {
        const float hv = h[c], pv = p[c];
        o0 += hv * sw[c*3+0] + pv * sw[192 + c*3+0];
        o1 += hv * sw[c*3+1] + pv * sw[192 + c*3+1];
        o2 += hv * sw[c*3+2] + pv * sw[192 + c*3+2];
    }
    Out[(size_t)t*3+0] = o0 + bias[n*3+0];
    Out[(size_t)t*3+1] = o1 + bias[n*3+1];
    Out[(size_t)t*3+2] = o2 + bias[n*3+2];
}

// =============================== host side =================================
static const int NUM_NODES_H[9] = {6890,3445,1723,862,431,216,108,54,27};
static const int RES_DIM_H[9]   = {64,64,128,128,256,256,512,512,512};

// act: 0 = plain, 1 = leaky(0.2)+bias (fc1 only)
static inline void launch_gemm(hipStream_t s, const float* A, const float* W,
                               const float* Cin, const float* bias, float* Out,
                               int M, int K, int N, int act)
{
    const int tilesN = N / 32;
    const int mTiles = M / 16;
    const int blocks = tilesN * ((mTiles + GEMM_WAVES - 1) / GEMM_WAVES);
    if (act)
        gemm_bf16_wmma<true, false, true><<<blocks, 256, 0, s>>>(A, W, nullptr, bias, Out, M, K, N);
    else if (Cin)
        gemm_bf16_wmma<false, true, false><<<blocks, 256, 0, s>>>(A, W, Cin, nullptr, Out, M, K, N);
    else
        gemm_bf16_wmma<false, false, false><<<blocks, 256, 0, s>>>(A, W, nullptr, nullptr, Out, M, K, N);
}

static inline void launch_zero(hipStream_t s, float* p, long long n)
{
    int blocks = (int)((n + 255) / 256);
    if (blocks > 16384) blocks = 16384;
    zero_kernel<<<blocks, 256, 0, s>>>(p, n);
}

extern "C" void kernel_launch(void* const* d_in, const int* in_sizes, int n_in,
                              void* d_out, int out_size, void* d_ws, size_t ws_size,
                              hipStream_t stream)
{
    (void)n_in; (void)out_size; (void)ws_size;
    #define FP(i) ((const float*)d_in[(i)])
    #define IP(i) ((const int*)  d_in[(i)])

    // ---- rebuild the setup_inputs() flattening order ----
    int idx = 0;
    const int i_x = idx++;
    const int i_fc1w = idx++, i_fc1b = idx++, i_conv1w = idx++;
    int bw1[8], bw2[8], bw3[8], bg1s[8], bg1b[8], bg2s[8], bg2b[8], bg3s[8], bg3b[8], bw4[8];
    for (int i = 0; i < 8; ++i) {
        bw1[i]=idx++; bw2[i]=idx++; bw3[i]=idx++;
        bg1s[i]=idx++; bg1b[i]=idx++; bg2s[i]=idx++; bg2b[i]=idx++; bg3s[i]=idx++; bg3b[i]=idx++;
        bw4[i] = (i==2 || i==4 || i==6) ? idx++ : -1;   // ci != co blocks
    }
    const int i_cow = idx++, i_obias = idx++;
    int gsrc[9], gdst[9], gnrm[9];
    for (int l = 0; l < 9; ++l) { gsrc[l]=idx++; gdst[l]=idx++; gnrm[l]=idx++; }
    int ucols[8], uvals[8];
    for (int i = 0; i < 8; ++i) { ucols[i]=idx++; uvals[i]=idx++; }

    // ---- workspace layout: 7 slots of max tensor (32*6890*64 floats) ----
    const size_t SLOT = (size_t)32 * 6890 * 64;     // 14,110,720 floats
    float* ws = (float*)d_ws;
    float* S0 = ws + 0 * SLOT;     // state ping
    float* S1 = ws + 1 * SLOT;     // state pong
    float* TA = ws + 2 * SLOT;     // scratch A
    float* TB = ws + 3 * SLOT;     // scratch B
    float* PP = ws + 4 * SLOT;     // propagation accumulator
    float* XU = ws + 5 * SLOT;     // pooled input
    float* RR = ws + 6 * SLOT;     // skip-path GEMM result

    // ---- head: fc1 (bias + leaky 0.2) then conv1 (K=1) ----
    // h = leaky(x @ fc1_w + b): M=32, K=128, N=1728  -> TA (== (32,27,64))
    launch_gemm(stream, FP(i_x), FP(i_fc1w), nullptr, FP(i_fc1b), TA, 32, 128, 1728, 1);
    // conv1: M=32*27=864, K=64, N=512 -> S0
    launch_gemm(stream, TA, FP(i_conv1w), nullptr, nullptr, S0, 864, 64, 512, 0);

    float* cur = S0;
    float* nxt = S1;

    // ---- 8 residual graph-conv blocks ----
    for (int i = 0; i < 8; ++i) {
        const int l  = 7 - i;
        const int nu = NUM_NODES_H[l];
        const int nd = NUM_NODES_H[l + 1];
        const int ci = RES_DIM_H[l + 1];
        const int co = RES_DIM_H[l];
        const int ch = co / 2;
        const int M  = BSZ * nu;
        const int E  = in_sizes[gsrc[l]];

        // xu = pool(x)
        {
            long long tot = (long long)BSZ * nu * ci;
            pool_kernel<<<(int)((tot + 255) / 256), 256, 0, stream>>>(
                cur, IP(ucols[i]), FP(uvals[i]), XU, nu, nd, ci);
        }
        // t = relu(gn1(xu))
        gn_relu_kernel<<<BSZ * 32, 256, 0, stream>>>(XU, FP(bg1s[i]), FP(bg1b[i]), TA, nu, ci);
        // h1 = t @ w1
        launch_gemm(stream, TA, FP(bw1[i]), nullptr, nullptr, TB, M, ci, ch, 0);
        // t = relu(gn2(h1))
        gn_relu_kernel<<<BSZ * 32, 256, 0, stream>>>(TB, FP(bg2s[i]), FP(bg2b[i]), TA, nu, ch);
        // cheb K=2: P = prop(t);  h2 = t@w2[0] + P@w2[1]
        launch_zero(stream, PP, (long long)M * ch);
        {
            long long tot = (long long)E * ch;
            prop_kernel<<<(int)((tot + 255) / 256), 256, 0, stream>>>(
                TA, PP, IP(gsrc[l]), IP(gdst[l]), FP(gnrm[l]), E, nu, ch);
        }
        launch_gemm(stream, TA, FP(bw2[i]),                   nullptr, nullptr, TB, M, ch, ch, 0);
        launch_gemm(stream, PP, FP(bw2[i]) + (size_t)ch * ch, TB,      nullptr, TB, M, ch, ch, 0);
        // t = relu(gn3(h2))
        gn_relu_kernel<<<BSZ * 32, 256, 0, stream>>>(TB, FP(bg3s[i]), FP(bg3b[i]), TA, nu, ch);
        // skip path
        const float* skip = XU;                       // ci == co: identity
        if (bw4[i] >= 0) {                            // ci != co: xu @ w4
            launch_gemm(stream, XU, FP(bw4[i]), nullptr, nullptr, RR, M, ci, co, 0);
            skip = RR;
        }
        // out = t @ w3 + skip
        launch_gemm(stream, TA, FP(bw3[i]), skip, nullptr, nxt, M, ch, co, 0);

        float* tmp = cur; cur = nxt; nxt = tmp;
    }

    // ---- tail: conv_out (K=2, 64 -> 3) on level 0 + out_bias ----
    {
        const int Nn = NUM_NODES_H[0];
        const int E0 = in_sizes[gsrc[0]];
        launch_zero(stream, PP, (long long)BSZ * Nn * 64);
        long long tot = (long long)E0 * 64;
        prop_kernel<<<(int)((tot + 255) / 256), 256, 0, stream>>>(
            cur, PP, IP(gsrc[0]), IP(gdst[0]), FP(gnrm[0]), E0, Nn, 64);
        conv_out_kernel<<<(BSZ * Nn + 255) / 256, 256, 0, stream>>>(
            cur, PP, FP(i_cow), FP(i_obias), (float*)d_out, Nn);
    }
    #undef FP
    #undef IP
}
